// ConditionalRandomField_45217415692495
// MI455X (gfx1250) — matrix operand order repaired and checked by reading, not compiled, not run on hardware
//
#include <hip/hip_runtime.h>

#define B_ 256
#define S_ 1024
#define T_ 128

#define PSTRIDE 136   // halfs per p-row (272B: 16B-aligned, +4 banks/row -> conflict-free b128)
#define CSTRIDE 132   // floats per score-row

#define LOG2E 1.44269504088896340736f
#define LN2   0.69314718055994530942f

typedef _Float16 v8h  __attribute__((ext_vector_type(8)));
typedef _Float16 v16h __attribute__((ext_vector_type(16)));
typedef float    v8f  __attribute__((ext_vector_type(8)));

// ---------------------------------------------------------------------------
// Kernel 1: numerator path score + per-batch mask sums.  One block per batch.
// ---------------------------------------------------------------------------
__global__ void __launch_bounds__(256)
crf_numer_kernel(const float* __restrict__ inp,
                 const int*   __restrict__ target,
                 const int*   __restrict__ mask,
                 const float* __restrict__ startT,
                 const float* __restrict__ endT,
                 const float* __restrict__ trans,
                 float* __restrict__ numer,
                 float* __restrict__ msumf) {
  __shared__ float red[256];
  __shared__ int   redi[256];
  const int b   = blockIdx.x;
  const int tid = threadIdx.x;
  const int*   tb = target + b * S_;
  const int*   mb = mask   + b * S_;
  const float* ib = inp + (size_t)b * S_ * T_;

  float s = 0.f;
  int   ms = 0;
  for (int t = tid; t < S_; t += 256) {
    const int mt = mb[t];
    ms += (mt > 0) ? 1 : 0;
    if (t >= 1 && mt > 0) {
      const int cur = tb[t], prev = tb[t - 1];
      s += trans[prev * T_ + cur] + ib[t * T_ + cur];
    }
  }
  red[tid] = s; redi[tid] = ms;
  __syncthreads();
  for (int off = 128; off > 0; off >>= 1) {
    if (tid < off) { red[tid] += red[tid + off]; redi[tid] += redi[tid + off]; }
    __syncthreads();
  }
  if (tid == 0) {
    const int totm = redi[0];
    const int t0   = tb[0];
    float sc = red[0] + startT[t0] + ib[t0];     // s=0 emission unconditional (matches ref)
    sc += endT[tb[totm - 1]];                    // last valid tag
    numer[b] = sc;
    msumf[b] = (float)totm;
  }
}

// ---------------------------------------------------------------------------
// Kernel 2: forward algorithm (normalizer) via exp-space WMMA matmul.
// grid = 16 blocks x 256 threads; block handles 16 batches; wave w = N-tile w.
// ---------------------------------------------------------------------------
__global__ void __launch_bounds__(256)
crf_forward_kernel(const float* __restrict__ inp,
                   const int*   __restrict__ mask,
                   const float* __restrict__ startT,
                   const float* __restrict__ endT,
                   const float* __restrict__ trans,
                   float* __restrict__ denom) {
  __shared__ __align__(16) float    sc[16][CSTRIDE];    // current log-scores
  __shared__ __align__(16) _Float16 pbuf[16][PSTRIDE];  // p = exp(sc - m), f16
  __shared__ __align__(16) float    mbuf[16];           // per-row max
  __shared__ __align__(16) int      mlds[S_][16];       // transposed mask [t][row]

  const int tid  = threadIdx.x;
  const int lane = tid & 31;
  const int wave = tid >> 5;            // 0..7 -> column tile
  const int b0   = blockIdx.x * 16;

  // phase-A mapping: 16 threads per row, 8 consecutive cols each
  const int arow = tid >> 4;
  const int acid = tid & 15;
  const int aj   = acid * 8;

  // phase-B per-lane geometry (WMMA layouts)
  const int nloc = lane & 15;           // column within tile / A-row
  const int hi   = lane >> 4;           // lane half
  const int jg   = wave * 16 + nloc;    // global tag column
  const int r0   = hi * 8;              // first owned output row

  // ---- stage mask transposed into LDS (coalesced global reads)
  for (int idx = tid; idx < 16 * S_; idx += 256) {
    const int row = idx >> 10;
    const int t   = idx & (S_ - 1);
    mlds[t][row] = mask[(b0 + row) * S_ + t];
  }

  // ---- B fragments: E = exp(transitions), contiguous k order (matches A):
  // lanes 0-15 hold K = kt*32+0..15 (elems 0..15); lanes 16-31 hold K = kt*32+16..31
  v16h bfrag[4];
  #pragma unroll
  for (int kt = 0; kt < 4; ++kt) {
    const int kb = kt * 32 + hi * 16;
    v16h bf;
    #pragma unroll
    for (int e = 0; e < 16; ++e)
      bf[e] = (_Float16)__expf(trans[(kb + e) * T_ + jg]);
    bfrag[kt] = bf;
  }

  // ---- t = 0 init: sc = start_transitions + emissions[:,0,:]
  {
    const float* row0 = inp + (size_t)(b0 + arow) * S_ * T_;
    #pragma unroll
    for (int e = 0; e < 8; ++e)
      sc[arow][aj + e] = startT[aj + e] + row0[aj + e];
  }
  __syncthreads();

  // each lane owns C-layout elements (row = r0 + r, col = jg): keep in regs
  float scv[8];
  #pragma unroll
  for (int r = 0; r < 8; ++r) scv[r] = sc[r0 + r][jg];

  // strength-reduced bases for the serial loop
  const float* ep = inp + ((size_t)(b0 + r0) * S_ + 1) * T_ + jg;  // + r*S*T per elem
  const int*   mp = &mlds[1][r0];

  for (int t = 1; t < S_; ++t) {
    // prefetch this step's emissions + mask (overlaps phase A)
    float emit[8];
    #pragma unroll
    for (int r = 0; r < 8; ++r) emit[r] = ep[(size_t)r * (S_ * T_)];
    int mv[8];
    #pragma unroll
    for (int r = 0; r < 8; ++r) mv[r] = mp[r];

    // ---- phase A: row max, p = exp2((sc - m)*log2e) -> f16 LDS
    float x[8];
    float m = -3.0e38f;
    #pragma unroll
    for (int e = 0; e < 8; ++e) { x[e] = sc[arow][aj + e]; m = fmaxf(m, x[e]); }
    #pragma unroll
    for (int o = 8; o >= 1; o >>= 1) m = fmaxf(m, __shfl_xor(m, o, 16));
    v8h pv;
    #pragma unroll
    for (int e = 0; e < 8; ++e)
      pv[e] = (_Float16)__builtin_amdgcn_exp2f((x[e] - m) * LOG2E);
    *(v8h*)&pbuf[arow][aj] = pv;
    if (acid == 0) mbuf[arow] = m;
    __syncthreads();

    // ---- phase B: (16x128 p) @ (128x16 E-tile); two independent WMMA chains
    v8f acc0 = {0.f, 0.f, 0.f, 0.f, 0.f, 0.f, 0.f, 0.f};
    v8f acc1 = {0.f, 0.f, 0.f, 0.f, 0.f, 0.f, 0.f, 0.f};
    v16h afr[4];
    #pragma unroll
    for (int kt = 0; kt < 4; ++kt) {
      // 16-bit A layout (contiguous, matches B): lane row = nloc; elems 0-15 = K kt*32 + hi*16 + 0..15.
      const int kb = kt * 32 + hi * 16;
      const v8h alo = *(const v8h*)&pbuf[nloc][kb];
      const v8h ahi = *(const v8h*)&pbuf[nloc][kb + 8];
      afr[kt] = __builtin_shufflevector(alo, ahi,
          0, 1, 2, 3, 4, 5, 6, 7, 8, 9, 10, 11, 12, 13, 14, 15);
    }
    acc0 = __builtin_amdgcn_wmma_f32_16x16x32_f16(false, afr[0], false, bfrag[0],
                                                  (short)0, acc0, false, false);
    acc1 = __builtin_amdgcn_wmma_f32_16x16x32_f16(false, afr[1], false, bfrag[1],
                                                  (short)0, acc1, false, false);
    acc0 = __builtin_amdgcn_wmma_f32_16x16x32_f16(false, afr[2], false, bfrag[2],
                                                  (short)0, acc0, false, false);
    acc1 = __builtin_amdgcn_wmma_f32_16x16x32_f16(false, afr[3], false, bfrag[3],
                                                  (short)0, acc1, false, false);
    const v8f acc = acc0 + acc1;

    // ---- epilogue: next = m + ln2*log2(p@E) + emit, branchless masked update
    float mvv[8];
    #pragma unroll
    for (int r = 0; r < 8; ++r) mvv[r] = mbuf[r0 + r];
    #pragma unroll
    for (int r = 0; r < 8; ++r) {
      const float newv = fmaf(__builtin_amdgcn_logf(acc[r]), LN2, mvv[r] + emit[r]);
      scv[r] = (mv[r] > 0) ? newv : scv[r];
      sc[r0 + r][jg] = scv[r];
    }
    ep += T_;
    mp += 16;
    __syncthreads();
  }

  // ---- denominator: logsumexp_j(sc + end_transitions)
  {
    float x[8];
    float m = -3.0e38f;
    #pragma unroll
    for (int e = 0; e < 8; ++e) {
      x[e] = sc[arow][aj + e] + endT[aj + e];
      m = fmaxf(m, x[e]);
    }
    #pragma unroll
    for (int o = 8; o >= 1; o >>= 1) m = fmaxf(m, __shfl_xor(m, o, 16));
    float ssum = 0.f;
    #pragma unroll
    for (int e = 0; e < 8; ++e)
      ssum += __builtin_amdgcn_exp2f((x[e] - m) * LOG2E);
    #pragma unroll
    for (int o = 8; o >= 1; o >>= 1) ssum += __shfl_xor(ssum, o, 16);
    if (acid == 0)
      denom[b0 + arow] = fmaf(__builtin_amdgcn_logf(ssum), LN2, m);
  }
}

// ---------------------------------------------------------------------------
// Kernel 3: final reduction -> mean llh scalar.
// ---------------------------------------------------------------------------
__global__ void __launch_bounds__(256)
crf_final_kernel(const float* __restrict__ ws, float* __restrict__ out) {
  __shared__ float red[256];
  __shared__ float redm[256];
  const int tid = threadIdx.x;
  red[tid]  = ws[tid] - ws[256 + tid];   // denom - numer
  redm[tid] = ws[512 + tid];             // mask sum per batch
  __syncthreads();
  for (int off = 128; off > 0; off >>= 1) {
    if (tid < off) { red[tid] += red[tid + off]; redm[tid] += redm[tid + off]; }
    __syncthreads();
  }
  if (tid == 0) out[0] = red[0] / redm[0];
}

// ---------------------------------------------------------------------------
extern "C" void kernel_launch(void* const* d_in, const int* in_sizes, int n_in,
                              void* d_out, int out_size, void* d_ws, size_t ws_size,
                              hipStream_t stream) {
  const float* inp    = (const float*)d_in[0];
  const int*   target = (const int*)  d_in[1];
  const int*   mask   = (const int*)  d_in[2];
  const float* startT = (const float*)d_in[3];
  const float* endT   = (const float*)d_in[4];
  const float* trans  = (const float*)d_in[5];

  float* ws    = (float*)d_ws;
  float* denom = ws;          // [0, 256)
  float* numer = ws + 256;    // [256, 512)
  float* msum  = ws + 512;    // [512, 768)

  crf_numer_kernel<<<B_, 256, 0, stream>>>(inp, target, mask, startT, endT, trans,
                                           numer, msum);
  crf_forward_kernel<<<B_ / 16, 256, 0, stream>>>(inp, mask, startT, endT, trans,
                                                  denom);
  crf_final_kernel<<<1, 256, 0, stream>>>(ws, (float*)d_out);
}
